// SparseRNN_2259152797729
// MI455X (gfx1250) — matrix operand order repaired and checked by reading, not compiled, not run on hardware
//
#include <hip/hip_runtime.h>
#include <math.h>

typedef __bf16 bf16;
typedef __attribute__((ext_vector_type(16))) __bf16 v16bf;
typedef __attribute__((ext_vector_type(8)))  __bf16 v8bf;
typedef __attribute__((ext_vector_type(8)))  float  v8f;

#define SEQ   512
#define EMBD  512
#define HIDD  512
#define BATCH 64
#define K1    5120          // EMB + 9*HID
#define NCLS  1000
#define WLEN  256
#define NWG   128           // persistent grid: one 16x16 tile per WG (4 x 32 tiles)
#define NTHR  256           // 8 wave32
#define NWAVE 8

// ---------------- fragment loaders (CDNA5 wave32 16-bit WMMA layouts) ----------------
// A 16x32 bf16 (per-lane): lanes0-15 row M=lane : V0-3=K c0..c0+7,   V4-7=K c0+16..23
//                          lanes16-31 row M=l-16: V0-3=K c0+8..c0+15,V4-7=K c0+24..31
// caller pre-applies row*stride + c0 + sel*8; we read p[0..7] and p[16..23].
__device__ __forceinline__ v16bf load_a16(const bf16* p) {
  v8bf lo = *(const v8bf*)p;
  v8bf hi = *(const v8bf*)(p + 16);
  v16bf a;
#pragma unroll
  for (int i = 0; i < 8; ++i) { a[i] = lo[i]; a[8 + i] = hi[i]; }
  return a;
}

__device__ __forceinline__ v16bf cvt_a16(const float* p) {   // f32 source -> bf16 frag
  v16bf a;
#pragma unroll
  for (int i = 0; i < 8; ++i) { a[i] = (bf16)p[i]; a[8 + i] = (bf16)p[16 + i]; }
  return a;
}

// B 32x16 bf16 from N-major weights WT[N][K]: lane column n, khalf=sel:
// V0-3 = K c0+sel*8..+7, V4-7 = K c0+16+sel*8..+7 (caller pre-applies n*K + c0 + sel*8).
__device__ __forceinline__ v16bf load_b16(const bf16* p) {
  v8bf lo = *(const v8bf*)p;
  v8bf hi = *(const v8bf*)(p + 16);
  v16bf b;
#pragma unroll
  for (int i = 0; i < 8; ++i) { b[i] = lo[i]; b[8 + i] = hi[i]; }
  return b;
}

#define WMMA_BF16(A, B, C) \
  __builtin_amdgcn_wmma_f32_16x16x32_bf16(false, (A), false, (B), (short)0, (C), false, false)

// ---------------- per-wave K-partial of one 16x16 tile -------------------------------
// Chunk mapping: wave wv owns kc = 16*blk + 2*wv + j, blk=0..9, j=0..1  (20 chunks).
// Block 0 = embedding (f32, EMB0=true) or z1cur (bf16); blocks 1..9 = dilated ring
// slots with offset 2^(9-blk) ("largest offset first" per the reference's reversal).
// Two independent accumulator chains (j=0 / j=1) hide the WMMA->WMMA RAW hazard.
template <bool EMB0>
__device__ __forceinline__ void gemm_phase(
    int t, const int* __restrict__ x_t, const float* __restrict__ emb,
    const bf16* __restrict__ z0base, const bf16* __restrict__ ring,
    const bf16* __restrict__ wtn,  // wt + (n0+row)*K1
    int arow, int sel, int wv, int lane, float* __restrict__ red_row) {
  const int coff = wv * 64;                         // wave's K offset inside a block
  const bf16* wb = wtn + coff + sel * 8;            // single B base, imm offsets below
  const int aoff = arow * HIDD + coff + sel * 8;
  v8f acc0 = 0.f, acc1 = 0.f;

  if (EMB0) {
    const int token = x_t[arow * SEQ];
    const float* er = emb + (size_t)token * EMBD + coff + sel * 8;
    acc0 = WMMA_BF16(cvt_a16(er),      load_b16(wb),      acc0);
    acc1 = WMMA_BF16(cvt_a16(er + 32), load_b16(wb + 32), acc1);
  } else {
    const bf16* pz = z0base + aoff;
    acc0 = WMMA_BF16(load_a16(pz),      load_b16(wb),      acc0);
    acc1 = WMMA_BF16(load_a16(pz + 32), load_b16(wb + 32), acc1);
  }
#pragma unroll
  for (int blk = 1; blk <= 9; ++blk) {
    const int slot = (t - (1 << (9 - blk))) & (WLEN - 1);
    const bf16* pa = ring + (size_t)slot * (BATCH * HIDD) + aoff;
    acc0 = WMMA_BF16(load_a16(pa),      load_b16(wb + 512 * blk),      acc0);
    acc1 = WMMA_BF16(load_a16(pa + 32), load_b16(wb + 512 * blk + 32), acc1);
  }
#pragma unroll
  for (int v = 0; v < 8; ++v) red_row[v * 32 + lane] = acc0[v] + acc1[v];
}

// ---------------- device-wide barrier (monotonic counter, generation targets) -------
__device__ __forceinline__ void grid_barrier(unsigned* cnt, unsigned target) {
  __threadfence();            // release this WG's global stores to device scope
  __syncthreads();
  if (threadIdx.x == 0) {
    __hip_atomic_fetch_add(cnt, 1u, __ATOMIC_RELEASE, __HIP_MEMORY_SCOPE_AGENT);
    while (__hip_atomic_load(cnt, __ATOMIC_ACQUIRE, __HIP_MEMORY_SCOPE_AGENT) < target)
      __builtin_amdgcn_s_sleep(1);
  }
  __syncthreads();
  __threadfence();            // acquire: every wave invalidates before reading
}

// ---------------- prep: counter reset, bf16 transposed weights, ring-buffer zero ----
__global__ __launch_bounds__(256) void prep_kernel(
    const float* __restrict__ W1, const float* __restrict__ W2,
    const float* __restrict__ Wo,
    bf16* __restrict__ w1t, bf16* __restrict__ w2t, bf16* __restrict__ wot,
    unsigned* __restrict__ bufzero, unsigned nzero_words, unsigned* __restrict__ cnt) {
  size_t tid    = (size_t)blockIdx.x * blockDim.x + threadIdx.x;
  size_t stride = (size_t)gridDim.x * blockDim.x;
  if (tid == 0) *cnt = 0u;
  for (size_t i = tid; i < (size_t)HIDD * K1; i += stride) {
    size_t n = i / K1, k = i % K1;
    w1t[i] = (bf16)W1[k * HIDD + n];
    w2t[i] = (bf16)W2[k * HIDD + n];
  }
  for (size_t i = tid; i < (size_t)NCLS * HIDD; i += stride) {
    size_t n = i / HIDD, k = i % HIDD;
    wot[i] = (bf16)Wo[k * NCLS + n];
  }
  for (size_t i = tid; i < nzero_words; i += stride) bufzero[i] = 0u;
}

// ---------------- persistent sequential recurrence --------------------------------
__global__ __launch_bounds__(NTHR) void rnn_recurrence(
    const int* __restrict__ x, const float* __restrict__ emb,
    const float* __restrict__ b1v, const float* __restrict__ b2v,
    const bf16* __restrict__ w1t, const bf16* __restrict__ w2t,
    bf16* __restrict__ buf1, bf16* __restrict__ buf2,
    bf16* __restrict__ z1cur, bf16* __restrict__ z2cur,
    bf16* __restrict__ z2all, unsigned* __restrict__ cnt) {
  const int tid  = threadIdx.x;
  const int wv   = tid >> 5;
  const int lane = tid & 31;
  const int row  = lane & 15;       // A row / B column within tile
  const int sel  = lane >> 4;       // lane-half selector
  const int wg   = blockIdx.x;
  const int m0   = (wg & 3) << 4;   // 4 batch tiles
  const int n0   = (wg >> 2) << 4;  // 32 hidden tiles
  const int tg   = wg * NTHR + tid; // 0..32767 flat copier id
  const int arow = m0 + row;
  const bf16* w1tn = w1t + (size_t)(n0 + row) * K1;
  const bf16* w2tn = w2t + (size_t)(n0 + row) * K1;
  // C/D layout mapping for the epilogue thread tid
  const int cM = (tid >> 5) + ((tid >> 4) & 1) * 8;
  const int cN = tid & 15;
  const int eb = m0 + cM, en = n0 + cN;
  const float bias1 = b1v[en];           // hoisted out of the 512-step loop
  const float bias2 = b2v[en];

  __shared__ float red[NWAVE][NTHR];
  unsigned bar = 0;

  for (int t = 0; t < SEQ; ++t) {
    // ============ phase 1: retire z2(t-1) into ring, GEMM1 -> z1cur ============
    if (t > 0) buf2[(size_t)((t - 1) & 255) * (BATCH * HIDD) + tg] = z2cur[tg];
    gemm_phase<true>(t, x + t, emb, (const bf16*)nullptr, buf1, w1tn,
                     arow, sel, wv, lane, red[wv]);
    __syncthreads();
    {
      float s = 0.f;
#pragma unroll
      for (int w = 0; w < NWAVE; ++w) s += red[w][tid];
      z1cur[(size_t)eb * HIDD + en] = (bf16)tanhf(s + bias1);
    }
    ++bar; grid_barrier(cnt, NWG * bar);

    // ============ phase 2: publish z1 into ring, GEMM2 -> z2cur/z2all ==========
    buf1[(size_t)(t & 255) * (BATCH * HIDD) + tg] = z1cur[tg];
    gemm_phase<false>(t, x + t, emb, z1cur, buf2, w2tn,
                      arow, sel, wv, lane, red[wv]);
    __syncthreads();
    {
      float s = 0.f;
#pragma unroll
      for (int w = 0; w < NWAVE; ++w) s += red[w][tid];
      bf16 zb = (bf16)tanhf(s + bias2);
      z2cur[(size_t)eb * HIDD + en] = zb;
      z2all[((size_t)t * BATCH + eb) * HIDD + en] = zb;  // staged for output GEMM
    }
    ++bar; grid_barrier(cnt, NWG * bar);
  }
}

// ---------------- fully parallel output GEMM: (512*64)x512 @ 512x1000 ---------------
__global__ __launch_bounds__(256) void out_gemm(
    const bf16* __restrict__ z2all, const bf16* __restrict__ wot,
    const float* __restrict__ bo, float* __restrict__ out) {
  const int tid  = threadIdx.x;
  const int wv   = tid >> 5;
  const int lane = tid & 31;
  const int row  = lane & 15;
  const int sel  = lane >> 4;
  const long tile = (long)blockIdx.x * 8 + wv;   // 2048 * 63 tiles, exact fit
  const int mt = (int)(tile / 63);
  const int nt = (int)(tile % 63);
  const int m0 = mt << 4;
  int nload = nt * 16 + row; if (nload > NCLS - 1) nload = NCLS - 1;  // clamp B column

  const bf16* pa = z2all + (size_t)(m0 + row) * HIDD + sel * 8;   // single A base
  const bf16* pb = wot + (size_t)nload * HIDD + sel * 8;          // single B base
  v8f acc0 = 0.f, acc1 = 0.f;
#pragma unroll
  for (int i = 0; i < 8; ++i) {                 // 16 K-chunks, two independent chains
    acc0 = WMMA_BF16(load_a16(pa + i * 64),      load_b16(pb + i * 64),      acc0);
    acc1 = WMMA_BF16(load_a16(pa + i * 64 + 32), load_b16(pb + i * 64 + 32), acc1);
  }
#pragma unroll
  for (int v = 0; v < 8; ++v) {
    const int tb = m0 + v + sel * 8;      // row index = t*64 + b
    const int tt = tb >> 6, bb = tb & 63;
    const int nn = nt * 16 + (lane & 15); // C layout: lane -> N
    if (nn < NCLS)
      out[((size_t)bb * SEQ + tt) * NCLS + nn] = acc0[v] + acc1[v] + bo[nn];
  }
}

// ---------------- host-side launcher -------------------------------------------------
extern "C" void kernel_launch(void* const* d_in, const int* in_sizes, int n_in,
                              void* d_out, int out_size, void* d_ws, size_t ws_size,
                              hipStream_t stream) {
  const int*   x   = (const int*)  d_in[0];
  const float* emb = (const float*)d_in[1];
  const float* W1  = (const float*)d_in[2];
  const float* b1  = (const float*)d_in[3];
  const float* W2  = (const float*)d_in[4];
  const float* b2  = (const float*)d_in[5];
  const float* Wo  = (const float*)d_in[6];
  const float* bo  = (const float*)d_in[7];
  float* out = (float*)d_out;

  // workspace layout (all 256B aligned); total ~78.8 MB
  char* ws = (char*)d_ws;
  unsigned* cnt = (unsigned*)ws;
  bf16* w1t  = (bf16*)(ws + 256);
  bf16* w2t  = w1t + (size_t)HIDD * K1;              // 2,621,440 elems
  bf16* wot  = w2t + (size_t)HIDD * K1;
  bf16* buf1 = wot + (size_t)NCLS * HIDD;            // 512,000 elems
  bf16* buf2 = buf1 + (size_t)WLEN * BATCH * HIDD;   // 8,388,608 elems each
  bf16* z1c  = buf2 + (size_t)WLEN * BATCH * HIDD;
  bf16* z2c  = z1c + (size_t)BATCH * HIDD;
  bf16* z2a  = z2c + (size_t)BATCH * HIDD;           // 16,777,216 elems

  const unsigned nzero_words = (2u * WLEN * BATCH * HIDD * 2u) / 4u;  // buf1+buf2
  prep_kernel<<<2048, 256, 0, stream>>>(W1, W2, Wo, w1t, w2t, wot,
                                        (unsigned*)buf1, nzero_words, cnt);

  rnn_recurrence<<<NWG, NTHR, 0, stream>>>(x, emb, b1, b2, w1t, w2t,
                                           buf1, buf2, z1c, z2c, z2a, cnt);

  // 2048 M-tiles * 63 N-tiles = 129024 tiles = 16128 WGs * 8 waves
  out_gemm<<<16128, 256, 0, stream>>>(z2a, wot, bo, out);
}
